// DifferentiableRollout_66185446031771
// MI455X (gfx1250) — compile-verified, matrix operand order checked by reading, not run on hardware
//
#include <hip/hip_runtime.h>

// Problem constants (match reference)
#define DT_F 0.1f
#define B_   1024
#define T_   200
#define SD_  64
#define CD_  32
#define H_   512
#define IN_  96   // SD + CD

typedef __attribute__((ext_vector_type(16))) __bf16 v16bf;
typedef __attribute__((ext_vector_type(8)))  float  v8f;

union Frag16 {
  v16bf v;
  int4  q[2];
  unsigned int u[8];
};

// f32 -> bf16 bits, round-to-nearest-even
__device__ __forceinline__ unsigned int f2bf(float x) {
  unsigned int u = __float_as_uint(x);
  return (u + 0x7FFFu + ((u >> 16) & 1u)) >> 16;
}
__device__ __forceinline__ unsigned int packbf2(float a, float b) {
  return f2bf(a) | (f2bf(b) << 16);
}

// Inverse of the 16-bit WMMA fragment K-striping:
// dword j of lane l holds K pair kloc = 16*(j>>2) + 8*(l>>4) + 2*(j&3).
// Given (idx, kloc) return the lane / dword that holds it (bit0 of kloc ignored).
__device__ __forceinline__ int frag_lane(int idx, int kloc) {
  return (((kloc >> 3) & 1) << 4) | idx;
}
__device__ __forceinline__ int frag_j(int kloc) {
  return (((kloc >> 4) & 1) << 2) | ((kloc & 7) >> 1);
}

// Cheap tanh: clamp + v_exp_f32 + v_rcp_f32 (plenty of accuracy for bf16 operands)
__device__ __forceinline__ float fast_tanh(float x) {
  float xc = fminf(fmaxf(x, -15.0f), 15.0f);
  float e  = __expf(2.0f * xc);                       // v_mul + v_exp
  float r  = __builtin_amdgcn_rcpf(e + 1.0f);         // v_rcp_f32
  return __builtin_fmaf(-2.0f, r, 1.0f);
}

// LDS budget (bytes), all operand tiles stored FRAGMENT-MAJOR (frag[lane][8 dwords]):
//  sW1f: 96 frags  *1KB = 98304   (W1 bf16: [kt 0..2][ntile 0..31])
//  sW2f: 64 frags  *1KB = 65536   (W2 bf16: [kt 0..15][ntile 0..3])
//  sb1 : 2048,  sb2 : 256
//  sX  : 4096   (fp32 master state [16][64])
//  sRed: 4096   (GEMM2 K-split partials [4][16][16])
//  sAf : 3 frags *1KB = 3072      (bf16 [x,u] A-tiles for GEMM1)
//  sHf : 16 frags*1KB = 16384     (bf16 hidden A-tiles for GEMM2)
#define SMEM_BYTES (98304 + 65536 + 2048 + 256 + 4096 + 4096 + 3072 + 16384)

__global__ __launch_bounds__(256, 1)
void rollout_wmma_kernel(const float* __restrict__ x0,
                         const float* __restrict__ controls,
                         const float* __restrict__ W1,
                         const float* __restrict__ b1,
                         const float* __restrict__ W2,
                         const float* __restrict__ b2,
                         float* __restrict__ out)
{
  extern __shared__ char smem[];
  unsigned int*   sW1f = (unsigned int*)smem;             // 96*256 u32
  unsigned int*   sW2f = sW1f + 96 * 256;                 // 64*256 u32
  float*          sb1  = (float*)(sW2f + 64 * 256);       // [512]
  float*          sb2  = sb1 + 512;                       // [64]
  float*          sX   = sb2 + 64;                        // [16][64]
  float*          sRed = sX + 16 * 64;                    // [4][16][16]
  unsigned int*   sAf  = (unsigned int*)(sRed + 4 * 16 * 16); // 3*256 u32
  unsigned int*   sHf  = sAf + 3 * 256;                   // 16*256 u32
  unsigned short* sHb  = (unsigned short*)sHf;

  const int tid  = threadIdx.x;
  const int w    = tid >> 5;     // wave id 0..7
  const int lane = tid & 31;
  const int half = lane >> 4;    // 0 or 1
  const int ln   = lane & 15;    // M (A-frag) or N (B/C/D-frag) index
  const int m0   = blockIdx.x * 16;

  const v8f vzero = {0.f, 0.f, 0.f, 0.f, 0.f, 0.f, 0.f, 0.f};

  // ---------------- init: stage weights/biases/x0 into LDS ----------------
  for (int i = tid; i < 96 * 256; i += 256) {       // W1 (96x512) -> fragment-major bf16
    int frag = i >> 8, r = i & 255;
    int l2 = r >> 3, j = r & 7;
    int kt = frag >> 5, nt = frag & 31;
    int kloc = ((j >> 2) << 4) + ((l2 >> 4) << 3) + ((j & 3) << 1);
    int k = kt * 32 + kloc, col = nt * 16 + (l2 & 15);
    sW1f[i] = packbf2(W1[k * H_ + col], W1[(k + 1) * H_ + col]);
  }
  for (int i = tid; i < 64 * 256; i += 256) {       // W2 (512x64) -> fragment-major bf16
    int frag = i >> 8, r = i & 255;
    int l2 = r >> 3, j = r & 7;
    int kt = frag >> 2, nt = frag & 3;
    int kloc = ((j >> 2) << 4) + ((l2 >> 4) << 3) + ((j & 3) << 1);
    int k = kt * 32 + kloc, col = nt * 16 + (l2 & 15);
    sW2f[i] = packbf2(W2[k * SD_ + col], W2[(k + 1) * SD_ + col]);
  }
  for (int i = tid; i < H_; i += 256) sb1[i] = b1[i];
  if (tid < SD_) sb2[tid] = b2[tid];

  { // x0 tile -> sX (fp32), sAf (bf16 frag-major), out[:,0,:]
    int base = tid * 4;                              // 1024 floats, 4/thread
    int m = base >> 6, c = base & 63;
    const float4 v = *(const float4*)&x0[(size_t)(m0 + m) * SD_ + c];
    *(float4*)&sX[base] = v;
    *(float4*)&out[(size_t)(m0 + m) * (T_ + 1) * SD_ + c] = v;
    int kt = c >> 5, kloc = c & 31;
    uint2 pk; pk.x = packbf2(v.x, v.y); pk.y = packbf2(v.z, v.w);
    *(uint2*)&sAf[kt * 256 + frag_lane(m, kloc) * 8 + frag_j(kloc)] = pk;
  }
  __syncthreads();

  // ---------------- recurrent rollout, fully on-chip ----------------
  for (int t = 0; t < T_; ++t) {
    { // controls(t) -> sAf frag kt=2 ; prefetch controls(t+1)
      int e = tid * 2;
      int m = e >> 5, c = e & 31;
      const float* cp = &controls[((size_t)(m0 + m) * T_ + t) * CD_ + c];
      const float2 u2 = *(const float2*)cp;
      sAf[2 * 256 + frag_lane(m, c) * 8 + frag_j(c)] = packbf2(u2.x, u2.y);
      if (t + 1 < T_) __builtin_prefetch(cp + CD_, 0, 1);  // global_prefetch_b8
    }
    __syncthreads();

    // ---- GEMM1: [16,96] x [96,512] ; wave w owns N-tiles [4w, 4w+4) ----
    {
      v8f acc[4];
      acc[0] = vzero; acc[1] = vzero; acc[2] = vzero; acc[3] = vzero;
      const int ntb = w * 4;
#pragma unroll
      for (int kt = 0; kt < 3; ++kt) {
        Frag16 a;
        const int ab = kt * 256 + lane * 8;
        a.q[0] = *(const int4*)&sAf[ab];
        a.q[1] = *(const int4*)&sAf[ab + 4];
#pragma unroll
        for (int nt = 0; nt < 4; ++nt) {
          Frag16 bm;
          const int fb = (kt * 32 + ntb + nt) * 256 + lane * 8;
          bm.q[0] = *(const int4*)&sW1f[fb];
          bm.q[1] = *(const int4*)&sW1f[fb + 4];
          acc[nt] = __builtin_amdgcn_wmma_f32_16x16x32_bf16(
              false, a.v, false, bm.v, (short)0, acc[nt], false, false);
        }
      }
      // bias + tanh, scatter bf16 h into GEMM2 fragment-major layout
#pragma unroll
      for (int nt = 0; nt < 4; ++nt) {
        const int hcol = (ntb + nt) * 16 + ln;       // K index for GEMM2
        const float bias = sb1[hcol];
        const int kk = hcol >> 5, kloc = hcol & 31;
        const int base16 = (kk * 256 + frag_j(kloc)) * 2 + (kloc & 1);
        const int h2 = ((kloc >> 3) & 1) << 4;       // target lane high bit
#pragma unroll
        for (int r = 0; r < 8; ++r) {
          const int m = r + 8 * half;                // C/D layout row
          const float hv = fast_tanh(acc[nt][r] + bias);
          sHb[base16 + (h2 | m) * 16] = (unsigned short)f2bf(hv);
        }
      }
    }
    __syncthreads();

    // ---- GEMM2: [16,512] x [512,64] ; wave = (N-tile w&3) x (K-half w>>2) ----
    const int nt2 = w & 3, kh = w >> 2;
    v8f acc2 = vzero;
#pragma unroll
    for (int kk = 0; kk < 8; ++kk) {
      const int kt2 = kh * 8 + kk;
      Frag16 a, bm;
      const int ab = kt2 * 256 + lane * 8;
      a.q[0] = *(const int4*)&sHf[ab];
      a.q[1] = *(const int4*)&sHf[ab + 4];
      const int fb = (kt2 * 4 + nt2) * 256 + lane * 8;
      bm.q[0] = *(const int4*)&sW2f[fb];
      bm.q[1] = *(const int4*)&sW2f[fb + 4];
      acc2 = __builtin_amdgcn_wmma_f32_16x16x32_bf16(
          false, a.v, false, bm.v, (short)0, acc2, false, false);
    }
    if (kh == 1) {     // upper-K partials to LDS
#pragma unroll
      for (int r = 0; r < 8; ++r)
        sRed[nt2 * 256 + (r + 8 * half) * 16 + ln] = acc2[r];
    }
    __syncthreads();

    if (kh == 0) {     // reduce + residual update in fp32
      const float bv = sb2[nt2 * 16 + ln];
#pragma unroll
      for (int r = 0; r < 8; ++r) {
        const int m = r + 8 * half;
        const int idx = m * SD_ + nt2 * 16 + ln;
        const float xn = sX[idx] +
            DT_F * (acc2[r] + sRed[nt2 * 256 + m * 16 + ln] + bv);
        sX[idx] = xn;
      }
    }
    __syncthreads();

    { // coalesced store of x_{t+1}; refresh bf16 A-frags for next step
      int base = tid * 4;
      int m = base >> 6, c = base & 63;
      const float4 v = *(const float4*)&sX[base];
      *(float4*)&out[(size_t)(m0 + m) * (T_ + 1) * SD_ + (size_t)(t + 1) * SD_ + c] = v;
      int kt = c >> 5, kloc = c & 31;
      uint2 pk; pk.x = packbf2(v.x, v.y); pk.y = packbf2(v.z, v.w);
      *(uint2*)&sAf[kt * 256 + frag_lane(m, kloc) * 8 + frag_j(kloc)] = pk;
    }
    __syncthreads();
  }
}

extern "C" void kernel_launch(void* const* d_in, const int* in_sizes, int n_in,
                              void* d_out, int out_size, void* d_ws, size_t ws_size,
                              hipStream_t stream) {
  (void)in_sizes; (void)n_in; (void)out_size; (void)d_ws; (void)ws_size;
  const float* x0       = (const float*)d_in[0];
  const float* controls = (const float*)d_in[1];
  const float* W1       = (const float*)d_in[2];
  const float* b1       = (const float*)d_in[3];
  const float* W2       = (const float*)d_in[4];
  const float* b2       = (const float*)d_in[5];
  float* out = (float*)d_out;

  dim3 grid(B_ / 16), block(256);
  rollout_wmma_kernel<<<grid, block, SMEM_BYTES, stream>>>(
      x0, controls, W1, b1, W2, b2, out);
}
